// Layer_Reduce_66700842107350
// MI455X (gfx1250) — matrix operand order repaired and checked by reading, not compiled, not run on hardware
//
#include <hip/hip_runtime.h>
#include <hip/hip_bf16.h>

typedef __attribute__((ext_vector_type(16))) _Float16 v16h;
typedef __attribute__((ext_vector_type(8)))  float    v8f;

#ifndef __has_builtin
#define __has_builtin(x) 0
#endif
#if __has_builtin(__builtin_amdgcn_global_load_async_to_lds_b128) && \
    __has_builtin(__builtin_amdgcn_s_wait_asynccnt)
#define HAVE_ASYNC_LDS 1
#else
#define HAVE_ASYNC_LDS 0
#endif

// 16-byte global -> LDS staging copy. CDNA5 path: GLOBAL_LOAD_ASYNC_TO_LDS_B128
// (ASYNCcnt-tracked, no VGPR round-trip). Fallback: b128 load + b128 ds store.
#if HAVE_ASYNC_LDS
typedef int i4 __attribute__((ext_vector_type(4)));
typedef __attribute__((address_space(1))) i4 gi4;  // global int4
typedef __attribute__((address_space(3))) i4 li4;  // LDS int4
static __device__ __forceinline__ void copy16(void* dst_lds, const void* src_glob) {
  __builtin_amdgcn_global_load_async_to_lds_b128((gi4*)src_glob, (li4*)dst_lds, 0, 0);
}
static __device__ __forceinline__ void async_join() {
  __builtin_amdgcn_s_wait_asynccnt(0);
}
#else
static __device__ __forceinline__ void copy16(void* dst_lds, const void* src_glob) {
  *(uint4*)dst_lds = *(const uint4*)src_glob;
}
static __device__ __forceinline__ void async_join() {}
#endif

// ---------------------------------------------------------------------------
// WMMA helpers (v_wmma_f32_16x16x32_f16). Layouts (A/B: contiguous k order chosen here, not the ISA table's):
//  A (16x32 f16): lane L holds row M=L&15; element e -> K = e + (L>=16?16:0)
//  B (32x16 f16): lane L holds col N=L&15; element e -> K = e + (L>=16?16:0)
//  C/D (16x16 f32): lane L col N=L&15; vgpr v -> row M = v + 8*(L>>4)
// All staged operands laid out so each lane reads CONTIGUOUS halves
// (vectorizes to ds_load_b128 pairs).
// ---------------------------------------------------------------------------
static __device__ __forceinline__ v8f wmma_f16(v16h a, v16h b, v8f c) {
  return __builtin_amdgcn_wmma_f32_16x16x32_f16(false, a, false, b, (short)0, c,
                                                false, false);
}

// A operand, row-major [row][k], row_stride in halves (multiple of 8).
static __device__ __forceinline__ v16h load_a_frag(const _Float16* base, int row,
                                                   int row_stride, int k0, int lane) {
  const _Float16* p = base + (size_t)row * row_stride + k0 + ((lane >= 16) ? 16 : 0);
  v16h a;
#pragma unroll
  for (int e = 0; e < 16; ++e) a[e] = p[e];
  return a;
}

// B operand staged TRANSPOSED: [col][k], col_stride in halves (multiple of 8).
static __device__ __forceinline__ v16h load_bT_frag(const _Float16* base, int col,
                                                    int col_stride, int k0, int lane) {
  const _Float16* p = base + (size_t)col * col_stride + k0 + ((lane >= 16) ? 16 : 0);
  v16h b;
#pragma unroll
  for (int e = 0; e < 16; ++e) b[e] = p[e];
  return b;
}

#define BATCH 8
#define CCH 256
#define NN 4096
#define SS 64

// ---------------------------------------------------------------------------
// K1: fold weights.  Wstack16 rows: [0,64)=W_qk, [64,128)=W_comb, [128,384)=W_v
//   W_comb = W_k2c[:, :64] @ W_qk  +  scatter(W_k2c[:, 64:], idx)
// Also convert W_t to f16.
// ---------------------------------------------------------------------------
__global__ __launch_bounds__(256) void k_prep(const float* __restrict__ W_qk,
                                              const float* __restrict__ W_k2c,
                                              const float* __restrict__ W_v,
                                              const float* __restrict__ W_t,
                                              const int* __restrict__ idx,
                                              _Float16* __restrict__ Wstack16,
                                              _Float16* __restrict__ Wt16) {
  const int g = blockIdx.x * 256 + threadIdx.x;
  const int NSTACK = 384 * 256;
  if (g < NSTACK) {
    int row = g >> 8, col = g & 255;
    float val;
    if (row < 64) {
      val = W_qk[row * 256 + col];
    } else if (row < 128) {
      int s = row - 64;
      val = 0.f;
      for (int j = 0; j < 64; ++j) val += W_k2c[s * 128 + j] * W_qk[j * 256 + col];
      for (int j = 0; j < 64; ++j)
        if (idx[j] == col) val += W_k2c[s * 128 + 64 + j];
    } else {
      val = W_v[(row - 128) * 256 + col];
    }
    Wstack16[g] = (_Float16)val;
  } else {
    int h = g - NSTACK;
    if (h < 256 * 256) Wt16[h] = (_Float16)W_t[h];
  }
}

// ---------------------------------------------------------------------------
// K2: per (b, 64-wide n tile): one stacked 384x256 GEMM against x tile.
//  rows 0..63   -> q16[b][n][s]
//  rows 64..127 -> column softmax over s -> kT16[b][n][s]   (k transposed)
//  rows 128..383-> + b_v -> xv16[b][c][n]
// ---------------------------------------------------------------------------
#define XS_STR 264  // 256 + 8 pad: 16B-aligned rows, spreads LDS banks

__global__ __launch_bounds__(256) void k_qkv(const float* __restrict__ x,
                                             const _Float16* __restrict__ Wstack16,
                                             const float* __restrict__ b_v,
                                             _Float16* __restrict__ q16,
                                             _Float16* __restrict__ kT16,
                                             _Float16* __restrict__ xv16) {
  const int b = blockIdx.y;
  const int n0 = blockIdx.x * 64;
  const int tid = threadIdx.x;
  const int lane = tid & 31;
  const int wave = tid >> 5;
  const int lc = lane & 15;

  __shared__ _Float16 xs[64 * XS_STR];            // ~33KB, [n_local][c]  (B operand)
  __shared__ __align__(16) char smemB[24 * 1024]; // wslab (24KB) / kscr (16KB)
  _Float16* wslab = (_Float16*)smemB;             // [384][32] k-slab (A operand)
  float* kscr = (float*)smemB;                    // [64][64] kraw scratch (after GEMM)

  const float* xb = x + (size_t)b * CCH * NN + n0;
  for (int i = tid; i < 256 * 64; i += 256) {
    int c = i >> 6, nl = i & 63;                  // coalesced global read along nl
    xs[nl * XS_STR + c] = (_Float16)xb[(size_t)c * NN + nl];
  }

  const v8f vzero = {};
  v8f acc[12];
#pragma unroll
  for (int t = 0; t < 12; ++t) acc[t] = vzero;

  for (int kk = 0; kk < 8; ++kk) {
    __syncthreads();
    for (int i = tid; i < 1536; i += 256) {       // 384 rows x 64B, async to LDS
      int r = i >> 2, c4 = i & 3;
      copy16(wslab + r * 32 + c4 * 8, Wstack16 + r * 256 + kk * 32 + c4 * 8);
    }
    async_join();
    __syncthreads();
#pragma unroll
    for (int t = 0; t < 12; ++t) {
      int tile = wave + t * 8;          // 96 tiles: 24 row-tiles x 4 col-tiles
      int rt = tile >> 2, ct = tile & 3;
      v16h a = load_a_frag(wslab, rt * 16 + lc, 32, 0, lane);
      v16h bf = load_bT_frag(xs, ct * 16 + lc, XS_STR, kk * 32, lane);
      acc[t] = wmma_f16(a, bf, acc[t]);
    }
  }
  __syncthreads();

#pragma unroll
  for (int t = 0; t < 12; ++t) {
    int tile = wave + t * 8;
    int rt = tile >> 2, ct = tile & 3;
    int col = ct * 16 + lc;
#pragma unroll
    for (int v = 0; v < 8; ++v) {
      int rit = v + ((lane >> 4) << 3);
      int grow = rt * 16 + rit;
      float val = acc[t][v];
      if (grow < 64) {
        q16[((size_t)b * NN + n0 + col) * SS + grow] = (_Float16)val;
      } else if (grow < 128) {
        kscr[(grow - 64) * 64 + col] = val;
      } else {
        int c = grow - 128;
        xv16[(size_t)b * CCH * NN + (size_t)c * NN + n0 + col] =
            (_Float16)(val + b_v[c]);
      }
    }
  }
  __syncthreads();

  if (tid < 64) {  // column softmax over the 64 s-rows; write k TRANSPOSED [n][s]
    float mx = -1e30f;
    for (int s = 0; s < 64; ++s) mx = fmaxf(mx, kscr[s * 64 + tid]);
    float sum = 0.f;
    for (int s = 0; s < 64; ++s) sum += __expf(kscr[s * 64 + tid] - mx);
    float inv = 1.0f / sum;
    _Float16* kout = kT16 + ((size_t)b * NN + n0 + tid) * SS;
    for (int s = 0; s < 64; ++s)
      kout[s] = (_Float16)(__expf(kscr[s * 64 + tid] - mx) * inv);
  }
}

// ---------------------------------------------------------------------------
// K3: pass 1 — online softmax row stats of energy = q·kT (never stored).
// ---------------------------------------------------------------------------
#define KS_STR 72  // 64 + 8 pad

__global__ __launch_bounds__(256) void k_rowstats(const _Float16* __restrict__ q16,
                                                  const _Float16* __restrict__ kT16,
                                                  float* __restrict__ rowmax,
                                                  float* __restrict__ rowsum) {
  const int b = blockIdx.y;
  const int n0 = blockIdx.x * 32;
  const int tid = threadIdx.x;
  const int lane = tid & 31;
  const int wave = tid >> 5;
  const int lc = lane & 15;

  __shared__ _Float16 qs[32 * 64];       // [n_local][s]   (A operand)
  __shared__ _Float16 ks[64 * KS_STR];   // [m_local][s]   (B operand, transposed)
  __shared__ float es[32 * 64];          // energy tile dump
  __shared__ float rmax[32], rsum[32];

  const _Float16* qbase = q16 + ((size_t)b * NN + n0) * SS;
  copy16(qs + tid * 8, qbase + tid * 8);  // 4KB q tile, one b128 per thread
  if (tid < 32) { rmax[tid] = -1e30f; rsum[tid] = 0.f; }

  const v8f vzero = {};
  for (int m0 = 0; m0 < NN; m0 += 64) {
    __syncthreads();
    for (int i = tid; i < 512; i += 256) {  // 64 rows x 128B k tile
      int ml = i >> 3, ch = i & 7;
      copy16(ks + ml * KS_STR + ch * 8,
             kT16 + ((size_t)b * NN + m0 + ml) * SS + ch * 8);
    }
    async_join();
    __syncthreads();
    {
      int nt = wave & 1, mt = wave >> 1; // 2 x 4 energy tiles
      v8f e = vzero;
#pragma unroll
      for (int kk = 0; kk < 2; ++kk) {   // K = S = 64
        v16h a = load_a_frag(qs, nt * 16 + lc, 64, kk * 32, lane);
        v16h bb = load_bT_frag(ks, mt * 16 + lc, KS_STR, kk * 32, lane);
        e = wmma_f16(a, bb, e);
      }
      int col = mt * 16 + lc;
#pragma unroll
      for (int v = 0; v < 8; ++v) {
        int rit = v + ((lane >> 4) << 3);
        es[(nt * 16 + rit) * 64 + col] = e[v];
      }
    }
    __syncthreads();
    if (tid < 32) {  // online max/sum update, one thread per row
      float om = rmax[tid];
      float tm = om;
      for (int j = 0; j < 64; ++j) tm = fmaxf(tm, es[tid * 64 + j]);
      float s = rsum[tid] * __expf(om - tm);
      for (int j = 0; j < 64; ++j) s += __expf(es[tid * 64 + j] - tm);
      rmax[tid] = tm;
      rsum[tid] = s;
    }
  }
  __syncthreads();
  if (tid < 32) {
    rowmax[(size_t)b * NN + n0 + tid] = rmax[tid];
    rowsum[(size_t)b * NN + n0 + tid] = rsum[tid];
  }
}

// ---------------------------------------------------------------------------
// K4: pass 2 fused — per (b, 32-wide m tile):
//   loop n: recompute A tile = exp(q·k - rowmax)/rowsum; accumulate
//   colsum[m] and acc[c][m] += x_v·A;  then /(1e-9+colsum);  W_t GEMM;
//   +b_t; BN; ReLU; residual; single f32 store. Attention never hits HBM.
// ---------------------------------------------------------------------------
#define AB_STR 40   // 32 + 8 pad
#define XR_STR 264  // 256 + 8 pad

__global__ __launch_bounds__(256) void k_attn_out(
    const float* __restrict__ x, const _Float16* __restrict__ q16,
    const _Float16* __restrict__ kT16, const _Float16* __restrict__ xv16,
    const _Float16* __restrict__ Wt16, const float* __restrict__ rowmax,
    const float* __restrict__ rowsum, const float* __restrict__ b_t,
    const float* __restrict__ bn_gamma, const float* __restrict__ bn_beta,
    const float* __restrict__ bn_mean, const float* __restrict__ bn_var,
    float* __restrict__ out) {
  const int b = blockIdx.y;
  const int m0 = blockIdx.x * 32;
  const int tid = threadIdx.x;
  const int lane = tid & 31;
  const int wave = tid >> 5;
  const int lc = lane & 15;

  __shared__ _Float16 kb[32 * KS_STR];   // [m_local][s], resident whole block (B)
  __shared__ _Float16 qb[32 * 64];       // [n_local][s]                       (A)
  __shared__ _Float16 ab[32 * AB_STR];   // A tile [m_local][n_local]          (B)
  __shared__ __align__(16) char uvb[32 * XR_STR * 2];  // vb [c][n] / xrb [m][c]
  __shared__ float cs[32];               // column sums of A
  _Float16* vb = (_Float16*)uvb;         // [256][32] x_v tile (A operand)
  _Float16* xrb = (_Float16*)uvb;        // [32][XR_STR] xr (B operand), reuse

  {  // 32 rows x 128B k tile; wait folded into first iteration's join
    int ml = tid >> 3, ch = tid & 7;
    copy16(kb + ml * KS_STR + ch * 8,
           kT16 + ((size_t)b * NN + m0 + ml) * SS + ch * 8);
  }
  if (tid < 32) cs[tid] = 0.f;

  const v8f vzero = {};
  v8f acc[4];
#pragma unroll
  for (int p = 0; p < 4; ++p) acc[p] = vzero;

  const float* rmb = rowmax + (size_t)b * NN;
  const float* rsb = rowsum + (size_t)b * NN;

  for (int n0 = 0; n0 < NN; n0 += 32) {
    __syncthreads();
    copy16(qb + tid * 8, q16 + ((size_t)b * NN + n0) * SS + tid * 8);  // 4KB q
    for (int i = tid; i < 1024; i += 256) {                            // 16KB x_v
      int c = i >> 2, q4 = i & 3;
      copy16(vb + c * 32 + q4 * 8,
             xv16 + (size_t)b * CCH * NN + (size_t)c * NN + n0 + q4 * 8);
    }
    if ((tid & 63) == 0 && n0 + 32 < NN)  // hint next q tile into cache
      __builtin_prefetch(q16 + ((size_t)b * NN + n0 + 32) * SS, 0, 1);
    async_join();
    __syncthreads();

    if (wave < 4) {  // energy 32x32 tile -> A (exp-normalized), 2x2 wmma tiles
      int nt = wave & 1, mt = wave >> 1;
      v8f e = vzero;
#pragma unroll
      for (int kk = 0; kk < 2; ++kk) {
        v16h a = load_a_frag(qb, nt * 16 + lc, 64, kk * 32, lane);
        v16h bb = load_bT_frag(kb, mt * 16 + lc, KS_STR, kk * 32, lane);
        e = wmma_f16(a, bb, e);
      }
      int col = mt * 16 + lc;
#pragma unroll
      for (int v = 0; v < 8; ++v) {
        int nl = (nt * 16) + v + ((lane >> 4) << 3);
        float rm = rmb[n0 + nl];
        float rs = rsb[n0 + nl];
        ab[col * AB_STR + nl] = (_Float16)(__expf(e[v] - rm) / rs);  // [m][n]
      }
    }
    __syncthreads();
    if (tid < 32) {  // accumulate column sums of A (row-contiguous)
      float s = 0.f;
      for (int nl = 0; nl < 32; ++nl) s += (float)ab[tid * AB_STR + nl];
      cs[tid] += s;
    }
#pragma unroll
    for (int p = 0; p < 4; ++p) {  // acc[c][m] += x_v · A ; K = 32 n's
      int ctile = wave * 2 + (p & 1);
      int mtile = p >> 1;
      v16h a = load_a_frag(vb, ctile * 16 + lc, 32, 0, lane);
      v16h bb = load_bT_frag(ab, mtile * 16 + lc, AB_STR, 0, lane);
      acc[p] = wmma_f16(a, bb, acc[p]);
    }
  }
  __syncthreads();

  // scale by 1/(1e-9 + colsum); stash xr TRANSPOSED [m][c] in xrb (reuses vb)
#pragma unroll
  for (int p = 0; p < 4; ++p) {
    int ctile = wave * 2 + (p & 1);
    int mtile = p >> 1;
    int ml = mtile * 16 + lc;
    float scale = 1.0f / (1e-9f + cs[ml]);
#pragma unroll
    for (int v = 0; v < 8; ++v) {
      int c = ctile * 16 + v + ((lane >> 4) << 3);
      xrb[ml * XR_STR + c] = (_Float16)(acc[p][v] * scale);
    }
  }
  __syncthreads();

  // W_t GEMM (256x256 x 256x32) + bias + BN + ReLU + residual
#pragma unroll
  for (int p = 0; p < 4; ++p) {
    int otile = wave * 2 + (p & 1);
    int mtile = p >> 1;
    v8f o = vzero;
    for (int kk = 0; kk < 8; ++kk) {
      v16h a = load_a_frag(Wt16, otile * 16 + lc, 256, kk * 32, lane);
      v16h bb = load_bT_frag(xrb, mtile * 16 + lc, XR_STR, kk * 32, lane);
      o = wmma_f16(a, bb, o);
    }
    int mg = m0 + mtile * 16 + lc;
#pragma unroll
    for (int v = 0; v < 8; ++v) {
      int oc = otile * 16 + v + ((lane >> 4) << 3);
      float val = o[v] + b_t[oc];
      float inv = bn_gamma[oc] * rsqrtf(bn_var[oc] + 1e-5f);
      val = (val - bn_mean[oc]) * inv + bn_beta[oc];
      val = fmaxf(val, 0.f);
      size_t oi = (size_t)b * CCH * NN + (size_t)oc * NN + mg;
      out[oi] = x[oi] + val;
    }
  }
}

// ---------------------------------------------------------------------------
extern "C" void kernel_launch(void* const* d_in, const int* in_sizes, int n_in,
                              void* d_out, int out_size, void* d_ws, size_t ws_size,
                              hipStream_t stream) {
  (void)in_sizes; (void)n_in; (void)out_size; (void)ws_size;
  const float* x        = (const float*)d_in[0];
  const float* W_qk     = (const float*)d_in[1];
  const float* W_k2c    = (const float*)d_in[2];
  const float* W_v      = (const float*)d_in[3];
  const float* b_v      = (const float*)d_in[4];
  const float* W_t      = (const float*)d_in[5];
  const float* b_t      = (const float*)d_in[6];
  const float* bn_gamma = (const float*)d_in[7];
  const float* bn_beta  = (const float*)d_in[8];
  const float* bn_mean  = (const float*)d_in[9];
  const float* bn_var   = (const float*)d_in[10];
  const int*   idx      = (const int*)d_in[11];
  float* out = (float*)d_out;

  char* ws = (char*)d_ws;
  size_t off = 0;
  auto take = [&](size_t bytes) {
    char* p = ws + off;
    off = (off + bytes + 255) & ~(size_t)255;
    return p;
  };
  _Float16* Wstack16 = (_Float16*)take((size_t)384 * 256 * 2);
  _Float16* Wt16     = (_Float16*)take((size_t)256 * 256 * 2);
  _Float16* q16      = (_Float16*)take((size_t)BATCH * NN * SS * 2);
  _Float16* kT16     = (_Float16*)take((size_t)BATCH * NN * SS * 2);
  _Float16* xv16     = (_Float16*)take((size_t)BATCH * CCH * NN * 2);
  float*    rowmax   = (float*)take((size_t)BATCH * NN * 4);
  float*    rowsum   = (float*)take((size_t)BATCH * NN * 4);

  k_prep<<<dim3((384 * 256 + 256 * 256 + 255) / 256), dim3(256), 0, stream>>>(
      W_qk, W_k2c, W_v, W_t, idx, Wstack16, Wt16);
  k_qkv<<<dim3(NN / 64, BATCH), dim3(256), 0, stream>>>(x, Wstack16, b_v, q16, kT16, xv16);
  k_rowstats<<<dim3(NN / 32, BATCH), dim3(256), 0, stream>>>(q16, kT16, rowmax, rowsum);
  k_attn_out<<<dim3(NN / 32, BATCH), dim3(256), 0, stream>>>(
      x, q16, kT16, xv16, Wt16, rowmax, rowsum, b_t, bn_gamma, bn_beta, bn_mean,
      bn_var, out);
}
